// MambaBlock_70738111365400
// MI455X (gfx1250) — compile-verified, matrix operand order checked by reading, not run on hardware
//
#include <hip/hip_runtime.h>
#include <hip/hip_bf16.h>

typedef __attribute__((ext_vector_type(16))) _Float16 v16h;
typedef __attribute__((ext_vector_type(8)))  float    v8f;
typedef __attribute__((ext_vector_type(4)))  unsigned int v4u;
typedef __attribute__((ext_vector_type(8)))  int      v8i;
typedef __attribute__((ext_vector_type(4)))  int      v4i;

#define MTOK   32768
#define LSEQ   4096
#define DM     192
#define DI     384
#define DST    64
#define NH     6
#define HD     64
#define NPAD   960
#define NCH    16
#define CHK    256

#if __has_builtin(__builtin_amdgcn_tensor_load_to_lds)
#define USE_TDM 1
#else
#define USE_TDM 0
#endif

__device__ __forceinline__ int kof(int i, int lane) {
    // 16-bit A/B fragment K index for vector element i (ISA 7.12.2)
    return ((i & 8) ? 16 : 0) + ((lane >> 4) << 3) + (i & 7);
}

__device__ __forceinline__ v8f vzero() {
    v8f v; for (int i = 0; i < 8; i++) v[i] = 0.0f; return v;
}

__device__ __forceinline__ v8f wmma_f16(v16h a, v16h b, v8f c) {
    return __builtin_amdgcn_wmma_f32_16x16x32_f16(false, a, false, b, (short)0, c, false, false);
}

__device__ __forceinline__ float wred(float v) {
    for (int m = 16; m > 0; m >>= 1) v += __shfl_xor(v, m, 32);
    return v;
}

#if USE_TDM
// TDM 2-D tile load: tile (tile_d0 x tile_d1) elements of 2 bytes from
// row-major tensor (row stride = stride0 elements) into LDS at lds_off,
// padding +4 DWORDs after every 16 DWORDs (row of 32 f16 -> stride 40 f16).
__device__ __forceinline__ void tdm_load_2d(unsigned lds_off, const void* gptr,
                                            unsigned tensor_d0, unsigned tensor_d1,
                                            unsigned stride0,
                                            unsigned tile_d0, unsigned tile_d1) {
    unsigned long long ga = (unsigned long long)(uintptr_t)gptr;
    v4u g0;
    g0[0] = 1u;                                            // count=1, user mode
    g0[1] = lds_off;                                       // lds_addr
    g0[2] = (unsigned)(ga & 0xffffffffu);                  // global_addr lo
    g0[3] = (unsigned)((ga >> 32) & 0x01ffffffu) | (2u << 30); // addr hi | type=2
    // data_size=2B (code1) | pad_enable | pad_interval=16dw(code3) | pad_amount=4dw(code3)
    unsigned d0flags = (1u << 16) | (1u << 20) | (3u << 22) | (3u << 25);
    v8i g1;
    g1[0] = (int)d0flags;
    g1[1] = (int)((tensor_d0 & 0xffffu) << 16);            // tensor_dim0 lo16 @ [63:48]
    g1[2] = (int)((tensor_d0 >> 16) | ((tensor_d1 & 0xffffu) << 16));
    g1[3] = (int)((tensor_d1 >> 16) | (tile_d0 << 16));    // tile_dim0 @ [127:112]
    g1[4] = (int)(tile_d1 & 0xffffu);                      // tile_dim1; tile_dim2=0
    g1[5] = (int)stride0;                                  // tensor_dim0_stride lo32
    g1[6] = 0;
    g1[7] = 0;
    v4i g2 = {0, 0, 0, 0};
    v4i g3 = {0, 0, 0, 0};
#if __clang_major__ >= 23
    v8i g4 = {0, 0, 0, 0, 0, 0, 0, 0};
    __builtin_amdgcn_tensor_load_to_lds(g0, g1, g2, g3, g4, 0);
#else
    __builtin_amdgcn_tensor_load_to_lds(g0, g1, g2, g3, 0);
#endif
}
#endif

// ---------------- LayerNorm 1: x (B,C,HW) -> xflat (tok,C) + h1 ----------------
__global__ void k_ln1(const float* __restrict__ x, const float* __restrict__ w,
                      const float* __restrict__ bb, float* __restrict__ xflat,
                      float* __restrict__ h1) {
    int wid = threadIdx.x >> 5, lane = threadIdx.x & 31;
    int tok = blockIdx.x * 4 + wid;
    int b = tok >> 12, t = tok & 4095;
    float v[6]; float s = 0.f;
    for (int j = 0; j < 6; j++) {
        int c = lane + 32 * j;
        v[j] = x[((b * DM + c) << 12) + t];
        s += v[j];
    }
    s = wred(s);
    float mu = s * (1.0f / DM);
    float q = 0.f;
    for (int j = 0; j < 6; j++) { float d = v[j] - mu; q += d * d; }
    q = wred(q);
    float rs = rsqrtf(q * (1.0f / DM) + 1e-5f);
    for (int j = 0; j < 6; j++) {
        int c = lane + 32 * j;
        xflat[tok * DM + c] = v[j];
        h1[tok * DM + c] = (v[j] - mu) * rs * w[c] + bb[c];
    }
}

// ---------------- weight fp32 -> f16 with row zero-pad ----------------
__global__ void k_cvt_pad(const float* __restrict__ src, _Float16* __restrict__ dst,
                          int N, int K) {
    int tid = blockIdx.x * 256 + threadIdx.x;     // over Npad*K
    int r = tid / K, c = tid % K;
    dst[tid] = (r < N) ? (_Float16)src[r * K + c] : (_Float16)0.f;
}

// ---------------- build u (funky reshape permutation) as f16 ----------------
__global__ void k_u16(const float* __restrict__ h1, _Float16* __restrict__ u16) {
    int tid = blockIdx.x * 256 + threadIdx.x;     // M*192
    int b = tid / (LSEQ * DM);
    int rem = tid - b * (LSEQ * DM);              // = t*DM + c
    int t = rem / DM, c = rem - t * DM;
    int idx = c * LSEQ + t;
    int t2 = idx / DM, c2 = idx - t2 * DM;
    u16[tid] = (_Float16)h1[(b * LSEQ + t2) * DM + c2];
}

// ---------------- generic WMMA GEMM: out = A(MxK) @ W(NxK)^T ----------------
// block tile 64(M) x 64(N), 8 waves, each wave two 16x16 tiles, K step 32
// A/W tiles DMA'd into LDS by the Tensor Data Mover when available.
// mode: 0 = f32 store, 1 = f16 store, 2 = f16 bias+gelu, 3 = f32 bias
__global__ void k_gemm16(const _Float16* __restrict__ A, const _Float16* __restrict__ W,
                         const float* __restrict__ bias, void* __restrict__ out,
                         int K, int ldo, int mode) {
    __shared__ _Float16 As[64][40];
    __shared__ _Float16 Ws[64][40];
    int tid = threadIdx.x;
    int wv = tid >> 5, l = tid & 31;
    int tm = wv & 3, tn0 = wv >> 2;               // n tiles: tn0 and tn0+2
    int m0 = blockIdx.x * 64, n0 = blockIdx.y * 64;
    v8f acc0 = vzero(), acc1 = vzero();
    int ar = tid >> 2, acb = (tid & 3) * 8;
    for (int kk = 0; kk < K; kk += 32) {
#if USE_TDM
        if (wv == 0) {
            tdm_load_2d((unsigned)(uintptr_t)&As[0][0], &A[(size_t)m0 * K + kk],
                        (unsigned)K, 64u, (unsigned)K, 32u, 64u);
            __builtin_amdgcn_s_wait_tensorcnt(0);
        } else if (wv == 1) {
            tdm_load_2d((unsigned)(uintptr_t)&Ws[0][0], &W[(size_t)n0 * K + kk],
                        (unsigned)K, 64u, (unsigned)K, 32u, 64u);
            __builtin_amdgcn_s_wait_tensorcnt(0);
        }
#else
        {
            const _Float16* src = &A[(size_t)(m0 + ar) * K + kk + acb];
            for (int i = 0; i < 8; i++) As[ar][acb + i] = src[i];
            const _Float16* ws = &W[(size_t)(n0 + ar) * K + kk + acb];
            for (int i = 0; i < 8; i++) Ws[ar][acb + i] = ws[i];
            if (kk + 32 < K)
                __builtin_prefetch(&A[(size_t)(m0 + ar) * K + kk + 32 + acb], 0, 0);
        }
#endif
        __syncthreads();
        v16h a, b0, b1;
        for (int i = 0; i < 16; i++) {
            int kf = kof(i, l);
            a[i]  = As[tm * 16 + (l & 15)][kf];
            b0[i] = Ws[tn0 * 16 + (l & 15)][kf];
            b1[i] = Ws[(tn0 + 2) * 16 + (l & 15)][kf];
        }
        acc0 = wmma_f16(a, b0, acc0);
        acc1 = wmma_f16(a, b1, acc1);
        __syncthreads();
    }
    for (int half = 0; half < 2; half++) {
        v8f acc = half ? acc1 : acc0;
        int tn = tn0 + 2 * half;
        for (int rr = 0; rr < 8; rr++) {
            int gm = m0 + tm * 16 + rr + ((l >> 4) << 3);
            int gn = n0 + tn * 16 + (l & 15);
            size_t idx = (size_t)gm * ldo + gn;
            float v = acc[rr];
            if (mode == 0) {
                ((float*)out)[idx] = v;
            } else if (mode == 1) {
                ((_Float16*)out)[idx] = (_Float16)v;
            } else if (mode == 2) {
                v += bias[gn];
                v = 0.5f * v * (1.0f + erff(v * 0.70710678f));
                ((_Float16*)out)[idx] = (_Float16)v;
            } else {
                v += bias[gn];
                ((float*)out)[idx] = v;
            }
        }
    }
}

// ---------------- dt = softplus(proj_dt + dt_bias) ----------------
__global__ void k_dt(const _Float16* __restrict__ P16, const float* __restrict__ dtb,
                     float* __restrict__ dt) {
    int tid = blockIdx.x * 256 + threadIdx.x;     // M*NH
    int g = tid / NH, h = tid - g * NH;
    float a = (float)P16[(size_t)g * NPAD + 896 + h] + dtb[h];
    dt[tid] = (a > 20.f) ? a : log1pf(__expf(a));
}

// ---------------- depthwise causal conv + SiLU, split x/B/C ----------------
__global__ void k_conv(const _Float16* __restrict__ P16, const float* __restrict__ cw,
                       const float* __restrict__ cb, const float* __restrict__ dt,
                       _Float16* __restrict__ xs16, _Float16* __restrict__ xdt16,
                       _Float16* __restrict__ B16, _Float16* __restrict__ C16) {
    int tid = blockIdx.x * 256 + threadIdx.x;     // M*512
    int g = tid >> 9, c = tid & 511;
    int b = g >> 12, t = g & 4095;
    float acc = cb[c];
    for (int k = 0; k < 4; k++) {
        int tt = t - 3 + k;
        if (tt >= 0)
            acc += cw[c * 4 + k] * (float)P16[(size_t)((b << 12) + tt) * NPAD + DI + c];
    }
    float s = acc / (1.0f + __expf(-acc));
    if (c < DI) {
        int h = c >> 6;
        xs16[(size_t)g * DI + c]  = (_Float16)s;
        xdt16[(size_t)g * DI + c] = (_Float16)(s * dt[g * NH + h]);
    } else if (c < DI + DST) {
        B16[(size_t)g * DST + (c - DI)] = (_Float16)s;
    } else {
        C16[(size_t)g * DST + (c - DI - DST)] = (_Float16)s;
    }
}

// ---------------- per-chunk inclusive cumsum of A = dt * (-exp(A_log)) ----------------
// Acs layout: [b][h][ch][256]
__global__ void k_cumsum(const float* __restrict__ dt, const float* __restrict__ alog,
                         float* __restrict__ Acs) {
    int cid = blockIdx.x;                          // B*NH*NCH = 768
    int b = cid / 96, rem = cid - b * 96;
    int h = rem / 16, ch = rem - h * 16;
    int i = threadIdx.x;
    int g = b * LSEQ + ch * CHK + i;
    float a = dt[g * NH + h] * (-__expf(alog[h]));
    __shared__ float sb[CHK];
    sb[i] = a;
    __syncthreads();
    for (int off = 1; off < CHK; off <<= 1) {
        float v = (i >= off) ? sb[i - off] : 0.f;
        __syncthreads();
        sb[i] += v;
        __syncthreads();
    }
    Acs[(size_t)cid * CHK + i] = sb[i];
}

// ---------------- per-chunk state: S(64p x 64n) = sum_j x[j,p]*decay[j]*B[j,n] ----------------
// cstate layout: [b][ch][h][64*64]
__global__ void k_cstate(const _Float16* __restrict__ xdt16, const _Float16* __restrict__ B16,
                         const float* __restrict__ Acs, float* __restrict__ cstate) {
    int cid = blockIdx.x;                          // 768 : [b][ch][h]
    int b = cid / 96, rem = cid - b * 96;
    int ch = rem / 6, h = rem - ch * 6;
    __shared__ float dec[CHK];
    int tid = threadIdx.x;
    size_t abase = ((size_t)(b * NH + h) * NCH + ch) * CHK;
    float last = Acs[abase + CHK - 1];
    dec[tid] = __expf(last - Acs[abase + tid]);
    __syncthreads();
    int wv = tid >> 5, l = tid & 31;
    int g0 = b * LSEQ + ch * CHK;
    size_t cbase = (size_t)cid * 4096;
    for (int ts = 0; ts < 2; ts++) {
        int tile = wv + 8 * ts, tm = tile >> 2, tn = tile & 3;
        int m = tm * 16 + (l & 15), n = tn * 16 + (l & 15);
        v8f acc = vzero();
        for (int kk = 0; kk < CHK; kk += 32) {
            v16h a, bf;
            for (int i = 0; i < 16; i++) {
                int j = kk + kof(i, l);
                a[i]  = (_Float16)((float)xdt16[(size_t)(g0 + j) * DI + h * HD + m] * dec[j]);
                bf[i] = B16[(size_t)(g0 + j) * DST + n];
            }
            acc = wmma_f16(a, bf, acc);
        }
        for (int rr = 0; rr < 8; rr++) {
            int ml = tm * 16 + rr + ((l >> 4) << 3);
            cstate[cbase + ml * 64 + n] = acc[rr];
        }
    }
}

// ---------------- inter-chunk state recurrence (sequential over 16 chunks) ----------------
__global__ void k_scan(const float* __restrict__ cstate, const float* __restrict__ Acs,
                       float* __restrict__ prior) {
    int bh = blockIdx.x;                           // B*NH = 48
    int b = bh / NH, h = bh - b * NH;
    int tid = threadIdx.x;
    for (int j = 0; j < 16; j++) {
        int e = tid + 256 * j;
        float pv = 0.f;
        for (int ch = 0; ch < NCH; ch++) {
            size_t idx = ((size_t)((b * NCH + ch) * NH + h)) * 4096 + e;
            prior[idx] = pv;
            float dsum = Acs[((size_t)(b * NH + h) * NCH + ch) * CHK + CHK - 1];
            pv = pv * __expf(dsum) + cstate[idx];
        }
    }
}

// ---------------- SSD core: Y = (C B^T o L) @ xdt + exp(Acs)*C@prior^T + D*xs ----------------
__global__ void k_y(const _Float16* __restrict__ C16, const _Float16* __restrict__ B16,
                    const _Float16* __restrict__ xdt16, const _Float16* __restrict__ xs16,
                    const float* __restrict__ Acs, const float* __restrict__ prior,
                    const float* __restrict__ Dsk, _Float16* __restrict__ y16) {
    int cid = blockIdx.x;                          // 768 : [b][ch][h]
    int b = cid / 96, rem = cid - b * 96;
    int ch = rem / 6, h = rem - ch * 6;
    __shared__ float acs[CHK];
    __shared__ _Float16 Sst[8][16][34];            // per-wave 16x32 score strip
    int tid = threadIdx.x;
    size_t abase = ((size_t)(b * NH + h) * NCH + ch) * CHK;
    acs[tid] = Acs[abase + tid];
    __syncthreads();
    int wv = tid >> 5, l = tid & 31;
    int g0 = b * LSEQ + ch * CHK;
    size_t pbase = (size_t)cid * 4096;
    float dskip = Dsk[h];
    for (int ss = 0; ss < 2; ss++) {
        int strip = wv * 2 + ss;
        int i0 = strip * 16;
        v8f accY[4];
        for (int pt = 0; pt < 4; pt++) accY[pt] = vzero();
        for (int jt2 = 0; jt2 < 8; jt2++) {
            if (2 * jt2 > strip) break;            // strictly upper-triangular pairs: zero
            // build masked exp-weighted score pair (16 x 32) into per-wave LDS
            for (int half = 0; half < 2; half++) {
                int jt = 2 * jt2 + half;
                if (jt > strip) {
                    for (int rr = 0; rr < 8; rr++)
                        Sst[wv][rr + ((l >> 4) << 3)][half * 16 + (l & 15)] = (_Float16)0.f;
                    continue;
                }
                v8f s = vzero();
                for (int kk = 0; kk < DST; kk += 32) {
                    v16h a, bf;
                    for (int i = 0; i < 16; i++) {
                        int ks = kk + kof(i, l);
                        a[i]  = C16[(size_t)(g0 + i0 + (l & 15)) * DST + ks];
                        bf[i] = B16[(size_t)(g0 + jt * 16 + (l & 15)) * DST + ks];
                    }
                    s = wmma_f16(a, bf, s);
                }
                for (int rr = 0; rr < 8; rr++) {
                    int il = i0 + rr + ((l >> 4) << 3);
                    int jl = jt * 16 + (l & 15);
                    float v = (jl <= il) ? s[rr] * __expf(acs[il] - acs[jl]) : 0.f;
                    Sst[wv][rr + ((l >> 4) << 3)][half * 16 + (l & 15)] = (_Float16)v;
                }
            }
            // Y += Sst(16x32) @ xdt(32 x 64)
            for (int pt = 0; pt < 4; pt++) {
                v16h a, bf;
                for (int i = 0; i < 16; i++) {
                    int kl = kof(i, l);
                    a[i] = Sst[wv][l & 15][kl];
                    int j = jt2 * 32 + kl;
                    bf[i] = xdt16[(size_t)(g0 + j) * DI + h * HD + pt * 16 + (l & 15)];
                }
                accY[pt] = wmma_f16(a, bf, accY[pt]);
            }
        }
        // inter-chunk: Y2[i,p] = sum_n C[i,n] * prior[p,n], scaled exp(acs[i])
        for (int pt = 0; pt < 4; pt++) {
            v8f acc2 = vzero();
            for (int kk = 0; kk < DST; kk += 32) {
                v16h a, bf;
                for (int i = 0; i < 16; i++) {
                    int ks = kk + kof(i, l);
                    a[i]  = C16[(size_t)(g0 + i0 + (l & 15)) * DST + ks];
                    bf[i] = (_Float16)prior[pbase + (pt * 16 + (l & 15)) * 64 + ks];
                }
                acc2 = wmma_f16(a, bf, acc2);
            }
            for (int rr = 0; rr < 8; rr++) {
                int il = i0 + rr + ((l >> 4) << 3);
                int p = pt * 16 + (l & 15);
                size_t gi = (size_t)(g0 + il);
                float yv = accY[pt][rr] + __expf(acs[il]) * acc2[rr]
                         + dskip * (float)xs16[gi * DI + h * HD + p];
                y16[gi * DI + h * HD + p] = (_Float16)yv;
            }
        }
    }
}

// ---------------- gating + RMSNorm: g = y*silu(z) * rsqrt(mean(g^2)) * w ----------------
__global__ void k_gate(const _Float16* __restrict__ y16, const _Float16* __restrict__ P16,
                       const float* __restrict__ nw, _Float16* __restrict__ g16) {
    int wid = threadIdx.x >> 5, l = threadIdx.x & 31;
    int g = blockIdx.x * 4 + wid;
    float gv[12]; float s = 0.f;
    for (int j = 0; j < 12; j++) {
        int c = l + 32 * j;
        float y = (float)y16[(size_t)g * DI + c];
        float z = (float)P16[(size_t)g * NPAD + c];
        float sz = z / (1.f + __expf(-z));
        gv[j] = y * sz;
        s += gv[j] * gv[j];
    }
    s = wred(s);
    float rs = rsqrtf(s * (1.0f / DI) + 1e-5f);
    for (int j = 0; j < 12; j++) {
        int c = l + 32 * j;
        g16[(size_t)g * DI + c] = (_Float16)(gv[j] * rs * nw[c]);
    }
}

// ---------------- residual add + LayerNorm2 -> h2 (f16) ----------------
__global__ void k_res_ln2(float* __restrict__ xflat, const float* __restrict__ mout,
                          const float* __restrict__ w, const float* __restrict__ bb,
                          _Float16* __restrict__ h2) {
    int wid = threadIdx.x >> 5, lane = threadIdx.x & 31;
    int tok = blockIdx.x * 4 + wid;
    float v[6]; float s = 0.f;
    for (int j = 0; j < 6; j++) {
        int c = lane + 32 * j;
        v[j] = xflat[tok * DM + c] + mout[tok * DM + c];
        s += v[j];
    }
    s = wred(s);
    float mu = s * (1.0f / DM);
    float q = 0.f;
    for (int j = 0; j < 6; j++) { float d = v[j] - mu; q += d * d; }
    q = wred(q);
    float rs = rsqrtf(q * (1.0f / DM) + 1e-5f);
    for (int j = 0; j < 6; j++) {
        int c = lane + 32 * j;
        xflat[tok * DM + c] = v[j];
        h2[(size_t)tok * DM + c] = (_Float16)((v[j] - mu) * rs * w[c] + bb[c]);
    }
}

// ---------------- final residual + output permutation (tok,C) -> (B,C,H,W) ----------------
__global__ void k_final(const float* __restrict__ xflat, const float* __restrict__ ffn,
                        float* __restrict__ out) {
    int tid = blockIdx.x * 256 + threadIdx.x;     // M*192
    int g = tid / DM, c = tid - g * DM;
    int b = g >> 12, t = g & 4095;
    out[((b * DM + c) << 12) + t] = xflat[tid] + ffn[tid];
}

extern "C" void kernel_launch(void* const* d_in, const int* in_sizes, int n_in,
                              void* d_out, int out_size, void* d_ws, size_t ws_size,
                              hipStream_t stream) {
    const float* x    = (const float*)d_in[0];
    const float* n1w  = (const float*)d_in[1];
    const float* n1b  = (const float*)d_in[2];
    const float* ipw  = (const float*)d_in[3];
    const float* cw   = (const float*)d_in[4];
    const float* cb   = (const float*)d_in[5];
    const float* dtb  = (const float*)d_in[6];
    const float* alog = (const float*)d_in[7];
    const float* dsk  = (const float*)d_in[8];
    const float* snw  = (const float*)d_in[9];
    const float* opw  = (const float*)d_in[10];
    const float* n2w  = (const float*)d_in[11];
    const float* n2b  = (const float*)d_in[12];
    const float* f1w  = (const float*)d_in[13];
    const float* f1b  = (const float*)d_in[14];
    const float* f2w  = (const float*)d_in[15];
    const float* f2b  = (const float*)d_in[16];
    float* out = (float*)d_out;

    char* ws = (char*)d_ws;
    size_t off = 0;
    auto alloc = [&](size_t bytes) -> char* {
        off = (off + 255) & ~(size_t)255;
        char* p = ws + off;
        off += bytes;
        return p;
    };
    const size_t M = MTOK;
    float*     xflat = (float*)alloc(M * DM * 4);
    char*      r1    = alloc(M * 768);            // h1(f32) then xdt16(f16)
    float*     h1    = (float*)r1;
    _Float16*  xdt16 = (_Float16*)r1;
    char*      r2    = alloc(M * DM * 2);         // u16 then h2_16
    _Float16*  u16   = (_Float16*)r2;
    _Float16*  h2_16 = (_Float16*)r2;
    _Float16*  w_ip  = (_Float16*)alloc((size_t)NPAD * DM * 2);
    _Float16*  w_op  = (_Float16*)alloc((size_t)DM * DI * 2);
    _Float16*  w_f1  = (_Float16*)alloc((size_t)DI * DM * 2);
    _Float16*  w_f2  = (_Float16*)alloc((size_t)DM * DI * 2);
    char*      rP    = alloc(M * NPAD * 2);       // P16; later mid16 + ffn
    _Float16*  P16   = (_Float16*)rP;
    _Float16*  mid16 = (_Float16*)rP;             // M*384 f16
    float*     ffn   = (float*)(rP + M * DI * 2); // M*192 f32
    float*     dt    = (float*)alloc(M * NH * 4);
    float*     Acs   = (float*)alloc((size_t)8 * NH * LSEQ * 4);
    _Float16*  xs16  = (_Float16*)alloc(M * DI * 2);
    _Float16*  B16   = (_Float16*)alloc(M * DST * 2);
    _Float16*  C16   = (_Float16*)alloc(M * DST * 2);
    float*     cstate= (float*)alloc((size_t)768 * 4096 * 4);
    float*     prior = (float*)alloc((size_t)768 * 4096 * 4);
    _Float16*  y16   = (_Float16*)alloc(M * DI * 2);
    _Float16*  g16   = (_Float16*)alloc(M * DI * 2);
    float*     mout  = (float*)alloc(M * DM * 4);

    // 1) LN1
    k_ln1<<<M / 4, 128, 0, stream>>>(x, n1w, n1b, xflat, h1);
    // 2) weight conversions
    k_cvt_pad<<<(NPAD * DM) / 256, 256, 0, stream>>>(ipw, w_ip, 902, DM);
    k_cvt_pad<<<(DM * DI) / 256, 256, 0, stream>>>(opw, w_op, DM, DI);
    k_cvt_pad<<<(DI * DM) / 256, 256, 0, stream>>>(f1w, w_f1, DI, DM);
    k_cvt_pad<<<(DM * DI) / 256, 256, 0, stream>>>(f2w, w_f2, DM, DI);
    // 3) build u (reshape permutation) in f16
    k_u16<<<(int)(M * DM / 256), 256, 0, stream>>>(h1, u16);
    // 4) in_proj GEMM -> P16 (M x 960, f16)
    k_gemm16<<<dim3(M / 64, NPAD / 64), 256, 0, stream>>>(u16, w_ip, nullptr, P16, DM, NPAD, 1);
    // 5) dt softplus
    k_dt<<<(int)(M * NH / 256), 256, 0, stream>>>(P16, dtb, dt);
    // 6) conv + silu + split
    k_conv<<<(int)(M * 512 / 256), 256, 0, stream>>>(P16, cw, cb, dt, xs16, xdt16, B16, C16);
    // 7) per-chunk cumsum of A
    k_cumsum<<<768, CHK, 0, stream>>>(dt, alog, Acs);
    // 8) per-chunk states (WMMA)
    k_cstate<<<768, 256, 0, stream>>>(xdt16, B16, Acs, cstate);
    // 9) inter-chunk recurrence
    k_scan<<<48, 256, 0, stream>>>(cstate, Acs, prior);
    // 10) SSD core (WMMA)
    k_y<<<768, 256, 0, stream>>>(C16, B16, xdt16, xs16, Acs, prior, dsk, y16);
    // 11) gating + RMSNorm
    k_gate<<<M / 4, 128, 0, stream>>>(y16, P16, snw, g16);
    // 12) out_proj GEMM -> mout (f32)
    k_gemm16<<<dim3(M / 64, DM / 64), 256, 0, stream>>>(g16, w_op, nullptr, mout, DI, DM, 0);
    // 13) residual + LN2 -> h2 (f16)
    k_res_ln2<<<M / 4, 128, 0, stream>>>(xflat, mout, n2w, n2b, h2_16);
    // 14) FFN1 GEMM (bias + exact GELU) -> mid16
    k_gemm16<<<dim3(M / 64, DI / 64), 256, 0, stream>>>(h2_16, w_f1, f1b, mid16, DM, DI, 2);
    // 15) FFN2 GEMM (bias) -> ffn (f32)
    k_gemm16<<<dim3(M / 64, DM / 64), 256, 0, stream>>>(mid16, w_f2, f2b, ffn, DI, DM, 3);
    // 16) final residual + output permutation
    k_final<<<(int)(M * DM / 256), 256, 0, stream>>>(xflat, ffn, out);
}
